// DecoderLayer_21921513079149
// MI455X (gfx1250) — matrix operand order repaired and checked by reading, not compiled, not run on hardware
//
#include <hip/hip_runtime.h>

// ---------------------------------------------------------------------------
// Decoder layer for MI455X (gfx1250, wave32, WMMA + Tensor Data Mover).
// All GEMMs: bf16 V_WMMA_F32_16X16X32_BF16, f32 accumulate.
// Global->LDS staging via TENSOR_LOAD_TO_LDS (TDM), double-buffered,
// synchronized with s_wait_tensorcnt + workgroup barriers.
// ---------------------------------------------------------------------------

typedef __bf16 bf16;
typedef bf16  v16bf __attribute__((ext_vector_type(16)));
typedef bf16  v8bf  __attribute__((ext_vector_type(8)));
typedef bf16  v4bf  __attribute__((ext_vector_type(4)));
typedef float v8f   __attribute__((ext_vector_type(8)));
typedef float v4f   __attribute__((ext_vector_type(4)));
typedef unsigned int v4u __attribute__((ext_vector_type(4)));
typedef int   v8i   __attribute__((ext_vector_type(8)));
typedef int   v4i   __attribute__((ext_vector_type(4)));

// Problem constants (fixed by the reference).
constexpr int CB = 8;      // batch
constexpr int CT = 1024;   // query length
constexpr int CA = 1024;   // kv length
constexpr int CC = 1024;   // channels
constexpr int CH = 16;     // heads
constexpr int CD = 64;     // head dim
constexpr int CF = 4096;   // FFN dim

// GEMM tiling: 256 threads = 8 waves. Block tile TM x BN_, K step 32.
// BN_=128: wave grid 4x2, wave tile 32x64 (2x4 WMMA tiles, 8 wmma/K-step).
// BN_=64 : wave grid 4x2, wave tile 32x32 (2x2 WMMA tiles)   [alpha @ V].
constexpr int TM = 128, TK = 32, NTHREADS = 256;
constexpr int LSTR = TK + 8;  // LDS row stride in bf16: 32 data + 8 pad (16B)

// Epilogue modes.
constexpr int M_BF16    = 0;  // bf16 plain (attention scores, with scale)
constexpr int M_BHTD    = 1;  // scatter bf16 to (B,H,T,D)  (q, k)
constexpr int M_BHDT    = 2;  // scatter bf16 to (B,H,D,T)  (v transposed)
constexpr int M_ATTN    = 3;  // scatter f32 to (B,T,C)     (alpha @ v)
constexpr int M_SELU    = 4;  // bias + SELU, bf16 plain    (FFN hidden)
constexpr int M_MASKRES = 5;  // (acc+bias)*rowmask + resid, f32 (final out)

// ---------------------------------------------------------------------------
// TDM: issue one 2D tile load (rows x 32 bf16) from global to LDS.
// LDS destination uses padded rows: 64B data + 16B pad  (= LSTR bf16).
// D# layout per CDNA5 ISA ch.8 (group0 128b, group1 256b).
// ---------------------------------------------------------------------------
__device__ __forceinline__ void tdm_load_tile(unsigned lds_off,
                                              const bf16* src,
                                              int rows, int stride_elems)
{
    unsigned long long ga = (unsigned long long)(uintptr_t)src;
    v4u g0;
    g0[0] = 1u;                                          // count=1, user D#
    g0[1] = lds_off;                                     // lds_addr (bytes)
    g0[2] = (unsigned)(ga & 0xffffffffu);                // global_addr[31:0]
    g0[3] = (unsigned)((ga >> 32) & 0x1ffffffu) | (2u << 30);  // [56:32]+type=2

    v8i g1;
    // data_size=1 (2B) @[17:16]; pad_enable @[20];
    // pad_interval=3 (16 DWORDs = 64B) @[24:22]; pad_amount=3 (4 DWORDs=16B) @[31:25]
    g1[0] = (1 << 16) | (1 << 20) | (3 << 22) | (3 << 25);
    g1[1] = (int)(32u << 16);                            // tensor_dim0[15:0]=32
    g1[2] = (int)(((unsigned)rows & 0xffffu) << 16);     // tensor_dim1[15:0]=rows
    g1[3] = (int)(32u << 16);                            // tile_dim0=32 @[127:112]
    g1[4] = (int)((unsigned)rows & 0xffffu);             // tile_dim1=rows; tile_dim2=0
    g1[5] = stride_elems;                                // tensor_dim0_stride[31:0]
    g1[6] = 0;                                           // stride hi / dim1_stride lo
    g1[7] = 0;

    v4i zz = {0, 0, 0, 0};
#if defined(__clang_major__) && (__clang_major__ >= 23)
    v8i z8 = {0, 0, 0, 0, 0, 0, 0, 0};
    __builtin_amdgcn_tensor_load_to_lds(g0, g1, zz, zz, z8, 0);
#else
    __builtin_amdgcn_tensor_load_to_lds(g0, g1, zz, zz, 0);
#endif
}

template <int MODE, int BN_, int WJ>
__global__ __launch_bounds__(NTHREADS) void gemm_wmma(
    const bf16* __restrict__ Ap, long long strideA,
    const bf16* __restrict__ Bp, long long strideB,
    void* __restrict__ Cp, long long strideC,
    int K, int lda, int ldb, int ldc,
    const float* __restrict__ bias, float scale,
    const float* __restrict__ rowmask, const float* __restrict__ resid)
{
    __shared__ bf16 sA[2][TM * LSTR];
    __shared__ bf16 sB[2][BN_ * LSTR];

    const int tid  = threadIdx.x;
    const int z    = blockIdx.z;
    const bf16* A  = Ap + (size_t)z * strideA;
    const bf16* B  = Bp + (size_t)z * strideB;
    const int m0   = blockIdx.y * TM;
    const int n0   = blockIdx.x * BN_;

    const int wave = tid >> 5;
    const int lane = tid & 31;
    const int wm   = (wave >> 1) * 32;        // wave M offset in block tile
    const int wn   = (wave & 1) * (WJ * 16);  // wave N offset in block tile
    const int fr   = lane & 15;               // row within 16-wide fragment
    const int hi   = lane >> 4;               // lane half select
    const int fka  = hi * 8;                  // A K sub-offset (0/8)
    const int fkb  = hi * 16;                 // unused (0/16); B fragments use fka

    v8f acc[2][WJ] = {};

    // Prologue: wave 0 kicks off DMA of the first K tile into buffer 0.
    if (wave == 0) {
        tdm_load_tile((unsigned)(uintptr_t)&sA[0][0], A + (size_t)m0 * lda, TM, lda);
        tdm_load_tile((unsigned)(uintptr_t)&sB[0][0], B + (size_t)n0 * ldb, BN_, ldb);
    }

    int cur = 0;
    for (int k0 = 0; k0 < K; k0 += TK, cur ^= 1) {
        if (wave == 0) {
            if (k0 + TK < K) {
                // Prefetch next K tile into the other buffer (its previous
                // readers all passed the trailing barrier of the last iter).
                tdm_load_tile((unsigned)(uintptr_t)&sA[cur ^ 1][0],
                              A + (size_t)m0 * lda + (k0 + TK), TM, lda);
                tdm_load_tile((unsigned)(uintptr_t)&sB[cur ^ 1][0],
                              B + (size_t)n0 * ldb + (k0 + TK), BN_, ldb);
                __builtin_amdgcn_s_wait_tensorcnt((short)2);  // current buf done
            } else {
                __builtin_amdgcn_s_wait_tensorcnt((short)0);
            }
        }
        __syncthreads();   // current buffer visible to all waves

        // A fragments: lane<16 -> K {0..7,16..23}; lane>=16 -> {8..15,24..31}
        v16bf af[2], bq[WJ];
#pragma unroll
        for (int i = 0; i < 2; ++i) {
            int r = wm + i * 16 + fr;
            union { v16bf v; v8bf h[2]; } u;
            u.h[0] = *(const v8bf*)&sA[cur][r * LSTR + fka];
            u.h[1] = *(const v8bf*)&sA[cur][r * LSTR + 16 + fka];
            af[i] = u.v;
        }
        // B fragments: lane<16 -> K {0..7,16..23} of col n; lane>=16 -> {8..15,24..31}
#pragma unroll
        for (int j = 0; j < WJ; ++j) {
            int r = wn + j * 16 + fr;
            union { v16bf v; v8bf h[2]; } u;
            u.h[0] = *(const v8bf*)&sB[cur][r * LSTR + fka];
            u.h[1] = *(const v8bf*)&sB[cur][r * LSTR + 16 + fka];
            bq[j] = u.v;
        }
#pragma unroll
        for (int i = 0; i < 2; ++i)
#pragma unroll
            for (int j = 0; j < WJ; ++j)
                acc[i][j] = __builtin_amdgcn_wmma_f32_16x16x32_bf16(
                    false, af[i], false, bq[j], (short)0, acc[i][j],
                    false, false);
        __syncthreads();   // all reads of current buffer done before re-DMA
    }

    // Epilogue. C/D layout: element e of acc -> M = e + 8*hi, N = lane&15.
#pragma unroll
    for (int i = 0; i < 2; ++i) {
#pragma unroll
        for (int j = 0; j < WJ; ++j) {
            int tm = m0 + wm + i * 16;
            int tn = n0 + wn + j * 16;
#pragma unroll
            for (int e = 0; e < 8; ++e) {
                int gm = tm + hi * 8 + e;
                int gn = tn + fr;
                float v = acc[i][j][e] * scale;
                if constexpr (MODE == M_BF16) {
                    bf16* C = (bf16*)Cp + (size_t)z * strideC;
                    C[(size_t)gm * ldc + gn] = (bf16)v;
                } else if constexpr (MODE == M_BHTD) {
                    int b = gm >> 10, t = gm & 1023;
                    int h = gn >> 6,  d = gn & 63;
                    ((bf16*)Cp)[(((size_t)(b * CH + h)) * CT + t) * CD + d] = (bf16)v;
                } else if constexpr (MODE == M_BHDT) {
                    int b = gm >> 10, t = gm & 1023;
                    int h = gn >> 6,  d = gn & 63;
                    ((bf16*)Cp)[(((size_t)(b * CH + h)) * CD + d) * CT + t] = (bf16)v;
                } else if constexpr (MODE == M_ATTN) {
                    int b = z >> 4, h = z & 15;   // z = b*H + h
                    ((float*)Cp)[((size_t)(b * CT + gm)) * CC + h * CD + gn] = v;
                } else if constexpr (MODE == M_SELU) {
                    v += bias[gn];
                    v = 1.0507009873554805f *
                        (v > 0.f ? v : 1.6732632423543772f * (__expf(v) - 1.f));
                    ((bf16*)Cp)[(size_t)gm * ldc + gn] = (bf16)v;
                } else {  // M_MASKRES
                    v += bias[gn];
                    v = v * rowmask[gm] + resid[(size_t)gm * ldc + gn];
                    ((float*)Cp)[(size_t)gm * ldc + gn] = v;
                }
            }
        }
    }
}

// f32 -> bf16 convert, 4 elements/thread.
__global__ __launch_bounds__(NTHREADS) void cvt_kernel(
    const float* __restrict__ in, bf16* __restrict__ out, int n4)
{
    int i = blockIdx.x * blockDim.x + threadIdx.x;
    if (i < n4) {
        v4f v = *(const v4f*)(in + (size_t)i * 4);
        v4bf o = {(bf16)v[0], (bf16)v[1], (bf16)v[2], (bf16)v[3]};
        *(v4bf*)(out + (size_t)i * 4) = o;
    }
}

// In-place masked softmax over one (b,h,t) row of bf16 scores (length 1024).
__global__ __launch_bounds__(NTHREADS) void softmax_kernel(
    bf16* __restrict__ scores, const unsigned char* __restrict__ mask)
{
    int idx = blockIdx.x;
    int t  = idx & (CT - 1);
    int bh = idx >> 10;
    int b  = bh >> 4;
    bf16* row = scores + ((size_t)bh * CT + t) * CA;
    const unsigned char* mrow = mask + ((size_t)(b * CT + t)) * CA;
    int tid = threadIdx.x;

    float x[4];
#pragma unroll
    for (int i = 0; i < 4; ++i) {
        int c = tid + i * NTHREADS;
        float s = (float)row[c];
        if (mrow[c]) s = -__builtin_inff();
        x[i] = s;
    }
    __shared__ float red[NTHREADS];
    float mx = fmaxf(fmaxf(x[0], x[1]), fmaxf(x[2], x[3]));
    red[tid] = mx; __syncthreads();
    for (int s = NTHREADS / 2; s > 0; s >>= 1) {
        if (tid < s) red[tid] = fmaxf(red[tid], red[tid + s]);
        __syncthreads();
    }
    mx = red[0]; __syncthreads();
    float sum = 0.f;
#pragma unroll
    for (int i = 0; i < 4; ++i) {
        float e = (x[i] == -__builtin_inff()) ? 0.f : __expf(x[i] - mx);
        x[i] = e; sum += e;
    }
    red[tid] = sum; __syncthreads();
    for (int s = NTHREADS / 2; s > 0; s >>= 1) {
        if (tid < s) red[tid] += red[tid + s];
        __syncthreads();
    }
    float inv = 1.0f / red[0];
#pragma unroll
    for (int i = 0; i < 4; ++i) {
        int c = tid + i * NTHREADS;
        row[c] = (bf16)(x[i] * inv);
    }
}

// out = LN(a + b) * gamma + beta, optional per-row mask; emits f32 + bf16.
__global__ __launch_bounds__(NTHREADS) void add_ln_kernel(
    const float* __restrict__ a, const float* __restrict__ b,
    const float* __restrict__ g, const float* __restrict__ be,
    const float* __restrict__ rowmask,
    float* __restrict__ outf, bf16* __restrict__ outb)
{
    int row = blockIdx.x;
    const float* ar = a + (size_t)row * CC;
    const float* br = b + (size_t)row * CC;
    int tid = threadIdx.x;

    float x[4]; float s = 0.f;
#pragma unroll
    for (int i = 0; i < 4; ++i) {
        int c = tid + i * NTHREADS;
        x[i] = ar[c] + br[c]; s += x[i];
    }
    __shared__ float red[NTHREADS];
    red[tid] = s; __syncthreads();
    for (int st = NTHREADS / 2; st > 0; st >>= 1) {
        if (tid < st) red[tid] += red[tid + st];
        __syncthreads();
    }
    float mean = red[0] * (1.0f / CC); __syncthreads();
    float vs = 0.f;
#pragma unroll
    for (int i = 0; i < 4; ++i) { float d = x[i] - mean; vs += d * d; }
    red[tid] = vs; __syncthreads();
    for (int st = NTHREADS / 2; st > 0; st >>= 1) {
        if (tid < st) red[tid] += red[tid + st];
        __syncthreads();
    }
    float rstd = rsqrtf(red[0] * (1.0f / CC) + 1e-5f);
    float rm = rowmask ? rowmask[row] : 1.0f;
#pragma unroll
    for (int i = 0; i < 4; ++i) {
        int c = tid + i * NTHREADS;
        float y = ((x[i] - mean) * rstd * g[c] + be[c]) * rm;
        outf[(size_t)row * CC + c] = y;
        outb[(size_t)row * CC + c] = (bf16)y;
    }
}

// ---------------------------------------------------------------------------
// Host side
// ---------------------------------------------------------------------------

template <int MODE, int BN_, int WJ>
static void launch_gemm(const bf16* A, long long sA, const bf16* B, long long sB,
                        void* C, long long sC, int M, int N, int K,
                        int lda, int ldb, int ldc,
                        const float* bias, float scale, int batch,
                        const float* rowm, const float* res, hipStream_t stream)
{
    dim3 grid(N / BN_, M / TM, batch), blk(NTHREADS);
    gemm_wmma<MODE, BN_, WJ><<<grid, blk, 0, stream>>>(
        A, sA, B, sB, C, sC, K, lda, ldb, ldc, bias, scale, rowm, res);
}

static void cvt(const float* in, bf16* out, size_t n, hipStream_t stream)
{
    int n4 = (int)(n / 4);
    cvt_kernel<<<(n4 + NTHREADS - 1) / NTHREADS, NTHREADS, 0, stream>>>(in, out, n4);
}

extern "C" void kernel_launch(void* const* d_in, const int* in_sizes, int n_in,
                              void* d_out, int out_size, void* d_ws, size_t ws_size,
                              hipStream_t stream)
{
    const float* x        = (const float*)d_in[0];
    const float* y        = (const float*)d_in[1];
    const unsigned char* ca_mask = (const unsigned char*)d_in[2];
    const unsigned char* sa_mask = (const unsigned char*)d_in[3];
    const float* mask     = (const float*)d_in[4];
    const float* sa_wq    = (const float*)d_in[5];
    const float* sa_wk    = (const float*)d_in[6];
    const float* sa_wv    = (const float*)d_in[7];
    const float* sa_g     = (const float*)d_in[8];
    const float* sa_b     = (const float*)d_in[9];
    const float* ca_wq    = (const float*)d_in[10];
    const float* ca_wk    = (const float*)d_in[11];
    const float* ca_wv    = (const float*)d_in[12];
    const float* ca_g     = (const float*)d_in[13];
    const float* ca_b     = (const float*)d_in[14];
    const float* w1       = (const float*)d_in[15];
    const float* b1       = (const float*)d_in[16];
    const float* w2       = (const float*)d_in[17];
    const float* b2       = (const float*)d_in[18];
    float* out = (float*)d_out;

    const size_t NTOK = (size_t)CB * CT;          // 8192 rows
    const size_t NELT = NTOK * CC;                // 8M activations

    char* ws = (char*)d_ws;
    size_t off = 0;
    auto alloc = [&](size_t bytes) -> void* {
        void* p = ws + off;
        off += (bytes + 255) & ~(size_t)255;
        return p;
    };
    bf16* actb   = (bf16*)alloc(NELT * 2);                 // activation bf16
    bf16* yb     = (bf16*)alloc(NELT * 2);                 // y bf16
    bf16* wb0    = (bf16*)alloc((size_t)CC * CC * 2);      // wq bf16
    bf16* wb1    = (bf16*)alloc((size_t)CC * CC * 2);      // wk bf16
    bf16* wb2    = (bf16*)alloc((size_t)CC * CC * 2);      // wv bf16
    bf16* w1b    = (bf16*)alloc((size_t)CF * CC * 2);
    bf16* w2b    = (bf16*)alloc((size_t)CC * CF * 2);
    bf16* qb     = (bf16*)alloc(NELT * 2);                 // (B,H,T,D)
    bf16* kb     = (bf16*)alloc(NELT * 2);                 // (B,H,T,D)
    bf16* vtb    = (bf16*)alloc(NELT * 2);                 // (B,H,D,T)
    bf16* scores = (bf16*)alloc((size_t)CB * CH * CT * CA * 2);  // 256 MB
    float* attn  = (float*)alloc(NELT * 4);
    float* resid = (float*)alloc(NELT * 4);
    bf16* h1b    = (bf16*)alloc(NTOK * (size_t)CF * 2);

    const float invTP = 0.08838834764831845f;  // 1/sqrt(2*D)
    const long long sQK = (long long)CT * CD;   // per-(b,h) q/k plane
    const long long sVT = (long long)CD * CT;   // per-(b,h) vT plane
    const long long sSC = (long long)CT * CA;   // per-(b,h) score plane
    const int BH = CB * CH;

    // ---- self-attention ----
    cvt(x, actb, NELT, stream);
    cvt(y, yb, NELT, stream);
    cvt(sa_wq, wb0, (size_t)CC * CC, stream);
    cvt(sa_wk, wb1, (size_t)CC * CC, stream);
    cvt(sa_wv, wb2, (size_t)CC * CC, stream);

    launch_gemm<M_BHTD, 128, 4>(actb, 0, wb0, 0, qb, 0, (int)NTOK, CC, CC, CC, CC, 0,
                                nullptr, 1.f, 1, nullptr, nullptr, stream);
    launch_gemm<M_BHTD, 128, 4>(actb, 0, wb1, 0, kb, 0, (int)NTOK, CC, CC, CC, CC, 0,
                                nullptr, 1.f, 1, nullptr, nullptr, stream);
    launch_gemm<M_BHDT, 128, 4>(actb, 0, wb2, 0, vtb, 0, (int)NTOK, CC, CC, CC, CC, 0,
                                nullptr, 1.f, 1, nullptr, nullptr, stream);

    launch_gemm<M_BF16, 128, 4>(qb, sQK, kb, sQK, scores, sSC, CT, CA, CD, CD, CD, CA,
                                nullptr, invTP, BH, nullptr, nullptr, stream);
    softmax_kernel<<<BH * CT, NTHREADS, 0, stream>>>(scores, sa_mask);
    launch_gemm<M_ATTN, 64, 2>(scores, sSC, vtb, sVT, attn, 0, CT, CD, CA, CA, CT, 0,
                               nullptr, 1.f, BH, nullptr, nullptr, stream);
    add_ln_kernel<<<(int)NTOK, NTHREADS, 0, stream>>>(
        attn, x, sa_g, sa_b, nullptr, resid, actb);

    // ---- cross-attention ----
    cvt(ca_wq, wb0, (size_t)CC * CC, stream);
    cvt(ca_wk, wb1, (size_t)CC * CC, stream);
    cvt(ca_wv, wb2, (size_t)CC * CC, stream);

    launch_gemm<M_BHTD, 128, 4>(actb, 0, wb0, 0, qb, 0, (int)NTOK, CC, CC, CC, CC, 0,
                                nullptr, 1.f, 1, nullptr, nullptr, stream);
    launch_gemm<M_BHTD, 128, 4>(yb, 0, wb1, 0, kb, 0, (int)NTOK, CC, CC, CC, CC, 0,
                                nullptr, 1.f, 1, nullptr, nullptr, stream);
    launch_gemm<M_BHDT, 128, 4>(yb, 0, wb2, 0, vtb, 0, (int)NTOK, CC, CC, CC, CC, 0,
                                nullptr, 1.f, 1, nullptr, nullptr, stream);

    launch_gemm<M_BF16, 128, 4>(qb, sQK, kb, sQK, scores, sSC, CT, CA, CD, CD, CD, CA,
                                nullptr, invTP, BH, nullptr, nullptr, stream);
    softmax_kernel<<<BH * CT, NTHREADS, 0, stream>>>(scores, ca_mask);
    launch_gemm<M_ATTN, 64, 2>(scores, sSC, vtb, sVT, attn, 0, CT, CD, CA, CA, CT, 0,
                               nullptr, 1.f, BH, nullptr, nullptr, stream);
    add_ln_kernel<<<(int)NTOK, NTHREADS, 0, stream>>>(
        attn, resid, ca_g, ca_b, mask, resid, actb);

    // ---- SELU FFN + masked residual ----
    cvt(w1, w1b, (size_t)CF * CC, stream);
    cvt(w2, w2b, (size_t)CC * CF, stream);
    launch_gemm<M_SELU, 128, 4>(actb, 0, w1b, 0, h1b, 0, (int)NTOK, CF, CC, CC, CC, CF,
                                b1, 1.f, 1, nullptr, nullptr, stream);
    launch_gemm<M_MASKRES, 128, 4>(h1b, 0, w2b, 0, out, 0, (int)NTOK, CC, CF, CF, CF, CC,
                                   b2, 1.f, 1, mask, resid, stream);
}
